// FastMultiHeadedAttention_4690104287900
// MI455X (gfx1250) — matrix operand order repaired and checked by reading, not compiled, not run on hardware
//
#include <hip/hip_runtime.h>

// ---------------- constants for this problem ----------------
#define BATCH   4
#define SEQ     2304            // 48*48
#define FDIM    512
#define NHEADS  8
#define DHEAD   64
#define MALL    (BATCH * SEQ)   // 9216
#define LDP     2312            // padded LDS row stride (floats) for logits

// ---------------- vector types (raw ext vectors: union-safe) ----------------
typedef __attribute__((ext_vector_type(16))) __bf16        v16bf;
typedef __attribute__((ext_vector_type(8)))  float         v8f;
typedef __attribute__((ext_vector_type(4)))  unsigned int  u32x4;
typedef __attribute__((ext_vector_type(4)))  float         f32x4;
typedef __attribute__((ext_vector_type(4)))  int           v4i;

#if defined(__has_builtin)
#  if __has_builtin(__builtin_amdgcn_global_load_async_to_lds_b128)
#    define HAVE_ASYNC_LDS 1
#  endif
#endif
#ifndef HAVE_ASYNC_LDS
#  define HAVE_ASYNC_LDS 0
#endif

union Frag {
  v16bf        v;     // operand for v_wmma_f32_16x16x32_bf16
  u32x4        q[2];  // two 16-byte chunks
  unsigned int d[8];  // packed bf16 pairs
};

// f32 -> bf16 (round-to-nearest-even), integer domain
__device__ __forceinline__ unsigned short f2bf(float x) {
  unsigned int u = __float_as_uint(x);
  u += 0x7FFFu + ((u >> 16) & 1u);
  return (unsigned short)(u >> 16);
}
__device__ __forceinline__ unsigned int pkbf(float lo, float hi) {
  return (unsigned int)f2bf(lo) | ((unsigned int)f2bf(hi) << 16);
}

__device__ __forceinline__ v8f wmma_bf16(const Frag& a, const Frag& b, v8f c) {
  return __builtin_amdgcn_wmma_f32_16x16x32_bf16(
      false, a.v, false, b.v, (short)0, c, false, false);
}

// A fragment (16x32 bf16): p = base + row*ld + (half<<3) + k0
__device__ __forceinline__ Frag ld_a_bf16(const unsigned short* p) {
  Frag f;
  f.q[0] = *(const u32x4*)(p);
  f.q[1] = *(const u32x4*)(p + 16);
  return f;
}
// B fragment (32x16 bf16) from B^T row-major: p = bt + col*ld + (half<<3) + k0
__device__ __forceinline__ Frag ld_b_bf16(const unsigned short* p) {
  Frag f;
  f.q[0] = *(const u32x4*)(p);
  f.q[1] = *(const u32x4*)(p + 16);
  return f;
}
// A fragment from f32 source (attention probs in LDS), convert on the fly
__device__ __forceinline__ Frag ld_a_f32(const float* p) {
  f32x4 x0 = *(const f32x4*)(p);
  f32x4 x1 = *(const f32x4*)(p + 4);
  f32x4 x2 = *(const f32x4*)(p + 16);
  f32x4 x3 = *(const f32x4*)(p + 20);
  Frag f;
  f.d[0] = pkbf(x0.x, x0.y); f.d[1] = pkbf(x0.z, x0.w);
  f.d[2] = pkbf(x1.x, x1.y); f.d[3] = pkbf(x1.z, x1.w);
  f.d[4] = pkbf(x2.x, x2.y); f.d[5] = pkbf(x2.z, x2.w);
  f.d[6] = pkbf(x3.x, x3.y); f.d[7] = pkbf(x3.z, x3.w);
  return f;
}

__device__ __forceinline__ void wait_async0() {
#if defined(__has_builtin) && __has_builtin(__builtin_amdgcn_s_wait_asynccnt)
  __builtin_amdgcn_s_wait_asynccnt(0);
#else
  asm volatile("s_wait_asynccnt 0x0" ::: "memory");
#endif
}

// ---------------- conversion kernels ----------------
__global__ __launch_bounds__(256) void cvt_f32_bf16(const float* __restrict__ src,
                                                    unsigned short* __restrict__ dst) {
  int i = blockIdx.x * 256 + threadIdx.x;     // one float4 per thread
  f32x4 f = ((const f32x4*)src)[i];
  ((unsigned int*)dst)[2 * i]     = pkbf(f.x, f.y);
  ((unsigned int*)dst)[2 * i + 1] = pkbf(f.z, f.w);
}

// W [F,F] f32 row-major -> W^T [F,F] bf16 (dst index contiguous)
__global__ __launch_bounds__(256) void cvt_w_transpose(const float* __restrict__ w,
                                                       unsigned short* __restrict__ wt) {
  int idx = blockIdx.x * 256 + threadIdx.x;   // 512*512 total
  int n = idx >> 9;
  int k = idx & 511;
  wt[idx] = f2bf(w[(k << 9) + n]);
}

// ---------------- projection GEMM: Y = A(bf16) @ B + bias ----------------
// MODE 0: bf16 out [b,h,S,d] (Q,K) | MODE 1: bf16 out [b,h,d,S] (V^T) | MODE 2: f32 out [MALL,512]
template <int MODE>
__global__ __launch_bounds__(256) void gemm_proj(const unsigned short* __restrict__ A,
                                                 const unsigned short* __restrict__ Bt,
                                                 const float* __restrict__ bias,
                                                 unsigned short* __restrict__ out_bf,
                                                 float* __restrict__ out_f32) {
  const int lane = threadIdx.x & 31;
  const int wave = threadIdx.x >> 5;
  const int m0   = (blockIdx.y * 8 + wave) << 4;
  const int n0   = blockIdx.x << 6;
  const int half = lane >> 4;
  const int lc   = lane & 15;

  const unsigned short* ap  = A  + (size_t)(m0 + lc) * FDIM + (half << 3);
  const unsigned short* bp0 = Bt + (size_t)(n0 + lc) * FDIM + (half << 3);

  v8f acc[4] = {};
  for (int k0 = 0; k0 < FDIM; k0 += 32) {
    Frag fa = ld_a_bf16(ap + k0);
#pragma unroll
    for (int j = 0; j < 4; ++j) {
      Frag fb = ld_b_bf16(bp0 + ((size_t)(j << 4) * FDIM) + k0);
      acc[j] = wmma_bf16(fa, fb, acc[j]);
    }
  }

  const int rbase = m0 + (half << 3);
  const int bidx  = rbase / SEQ;
  const int sbase = rbase - bidx * SEQ;
#pragma unroll
  for (int j = 0; j < 4; ++j) {
    const int n  = n0 + (j << 4) + lc;
    const float bv = bias[n];
    const int h = n >> 6, dd = n & 63;
#pragma unroll
    for (int i = 0; i < 8; ++i) {
      const float val = acc[j][i] + bv;
      const int s = sbase + i;
      if constexpr (MODE == 0) {
        out_bf[((size_t)((bidx * NHEADS + h) * SEQ + s) << 6) + dd] = f2bf(val);
      } else if constexpr (MODE == 1) {
        out_bf[(size_t)((bidx * NHEADS + h) * DHEAD + dd) * SEQ + s] = f2bf(val);
      } else {
        out_f32[((size_t)(rbase + i) << 9) + n] = val;
      }
    }
  }
}

// ---------------- fused attention: logits -> softmax -> attn-out -> ctx ----------------
// One block = one (b,h) and 16 query rows. Full 16x2304 logits row-block lives in LDS
// (CDNA5: 320KB/WGP). attn touches HBM exactly once (NT stores).
extern __shared__ char smem_raw[];

__global__ __launch_bounds__(256) void attn_fused(
    const unsigned short* __restrict__ Q,    // [bh,S,d] bf16
    const unsigned short* __restrict__ Kb,   // [bh,S,d] bf16
    const unsigned short* __restrict__ Vt,   // [bh,d,S] bf16
    float* __restrict__ attn,                // [bh,S,S] f32 (in d_out)
    unsigned short* __restrict__ Ctx) {      // [B*S,F] bf16
  const int bh   = blockIdx.y;
  const int b    = bh >> 3;
  const int h    = bh & 7;
  const int m0   = blockIdx.x << 4;
  const int tid  = threadIdx.x;
  const int lane = tid & 31;
  const int wave = tid >> 5;
  const int half = lane >> 4;
  const int lc   = lane & 15;

  float*          s_log = (float*)smem_raw;                              // 16 x LDP f32
  unsigned short* s_q   = (unsigned short*)(smem_raw + 16 * LDP * 4);    // 16 x 64 bf16
  float*          s_red = (float*)(smem_raw + 16 * LDP * 4 + 2048);      // 16 x 64 f32

  // ---- stage q tile (2 KB) into LDS (async on CDNA5) ----
  {
    const unsigned short* qh = Q + (size_t)bh * SEQ * DHEAD + (size_t)m0 * DHEAD;
#if HAVE_ASYNC_LDS
    if (tid < 128) {
      __builtin_amdgcn_global_load_async_to_lds_b128(
          (__attribute__((address_space(1))) v4i*)(qh + tid * 8),
          (__attribute__((address_space(3))) v4i*)(s_q + tid * 8),
          0, 0);
    }
    wait_async0();
#else
    if (tid < 128) {
      u32x4 t = *(const u32x4*)(qh + tid * 8);
      *(u32x4*)(s_q + tid * 8) = t;
    }
#endif
    __syncthreads();
  }

  // A fragments for this q tile (reused for all 144 N tiles)
  Frag qa0, qa1;
  {
    const unsigned short* qrow = s_q + lc * DHEAD + (half << 3);
    qa0 = ld_a_bf16(qrow);        // K 0..31
    qa1 = ld_a_bf16(qrow + 32);   // K 32..63
  }

  // ---- phase 1: logits into LDS ----
  {
    const unsigned short* kh = Kb + (size_t)bh * SEQ * DHEAD;
    for (int t = 0; t < 18; ++t) {                    // 8 waves x 18 tiles = 144
      const int n0 = ((wave * 18) + t) << 4;
      const unsigned short* bp = kh + (size_t)(n0 + lc) * DHEAD + (half << 3);
      v8f acc = {0.f, 0.f, 0.f, 0.f, 0.f, 0.f, 0.f, 0.f};
      acc = wmma_bf16(qa0, ld_b_bf16(bp), acc);
      acc = wmma_bf16(qa1, ld_b_bf16(bp + 32), acc);
      const int rbase = half << 3;
#pragma unroll
      for (int i = 0; i < 8; ++i)
        s_log[(rbase + i) * LDP + n0 + lc] = acc[i] * 0.125f;  // 1/sqrt(64)
    }
  }
  __syncthreads();

  // ---- phase 2: softmax per row + NT store of attn ----
  {
    float* arow_out = attn + (size_t)bh * SEQ * SEQ;
#pragma unroll
    for (int r2 = 0; r2 < 2; ++r2) {
      const int r = wave * 2 + r2;
      f32x4* row4 = (f32x4*)(s_log + r * LDP);
      float mx = -3.402823e38f;
      for (int k = 0; k < 18; ++k) {                 // 18*32*4 = 2304
        f32x4 v = row4[lane + 32 * k];
        mx = fmaxf(mx, fmaxf(fmaxf(v.x, v.y), fmaxf(v.z, v.w)));
      }
#pragma unroll
      for (int o = 16; o > 0; o >>= 1) mx = fmaxf(mx, __shfl_xor(mx, o, 32));
      float sum = 0.f;
      for (int k = 0; k < 18; ++k) {
        f32x4 v = row4[lane + 32 * k];
        v.x = __expf(v.x - mx); v.y = __expf(v.y - mx);
        v.z = __expf(v.z - mx); v.w = __expf(v.w - mx);
        row4[lane + 32 * k] = v;
        sum += (v.x + v.y) + (v.z + v.w);
      }
#pragma unroll
      for (int o = 16; o > 0; o >>= 1) sum += __shfl_xor(sum, o, 32);
      const float inv = 1.0f / sum;
      f32x4* grow4 = (f32x4*)(arow_out + (size_t)(m0 + r) * SEQ);
      for (int k = 0; k < 18; ++k) {
        f32x4 v = row4[lane + 32 * k];
        v *= inv;
        row4[lane + 32 * k] = v;                               // keep for ctx phase
        __builtin_nontemporal_store(v, &grow4[lane + 32 * k]); // write-once stream
      }
    }
  }
  __syncthreads();

  // ---- phase 3: ctx = attn(LDS f32) @ V, split-K across wave pairs ----
  {
    const int j     = wave & 3;        // N tile (16 of 64)
    const int khalf = wave >> 2;       // K half
    const unsigned short* vp = Vt + (size_t)bh * DHEAD * SEQ +
                               (size_t)((j << 4) + lc) * SEQ + (half << 3);
    const float* aprow = s_log + lc * LDP + (half << 3);
    v8f acc = {0.f, 0.f, 0.f, 0.f, 0.f, 0.f, 0.f, 0.f};
    const int kbeg = khalf * (SEQ / 2);
    for (int k0 = kbeg; k0 < kbeg + SEQ / 2; k0 += 32) {
      __builtin_prefetch(vp + k0 + 1024, 0, 1);      // global_prefetch on V stream
      Frag fa = ld_a_f32(aprow + k0);                // LDS f32 -> bf16 frag
      Frag fb = ld_b_bf16(vp + k0);
      acc = wmma_bf16(fa, fb, acc);
    }
    const int rbase = half << 3;
    if (khalf == 1) {
#pragma unroll
      for (int i = 0; i < 8; ++i)
        s_red[((rbase + i) << 6) + (j << 4) + lc] = acc[i];
    }
    __syncthreads();
    if (khalf == 0) {
#pragma unroll
      for (int i = 0; i < 8; ++i) {
        const int r = rbase + i;
        const float v = acc[i] + s_red[(r << 6) + (j << 4) + lc];
        const int s = m0 + r;
        Ctx[((size_t)(b * SEQ + s) << 9) + (h << 6) + (j << 4) + lc] = f2bf(v);
      }
    }
  }
}

// ---------------- host-side orchestration ----------------
extern "C" void kernel_launch(void* const* d_in, const int* in_sizes, int n_in,
                              void* d_out, int out_size, void* d_ws, size_t ws_size,
                              hipStream_t stream) {
  (void)in_sizes; (void)n_in; (void)out_size; (void)ws_size;

  const float* query = (const float*)d_in[0];
  const float* value = (const float*)d_in[1];
  const float* keys  = (const float*)d_in[2];
  const float* wq_w  = (const float*)d_in[3];
  const float* wq_b  = (const float*)d_in[4];
  const float* wk_w  = (const float*)d_in[5];
  const float* wk_b  = (const float*)d_in[6];
  const float* wv_w  = (const float*)d_in[7];
  const float* wv_b  = (const float*)d_in[8];
  const float* wo_w  = (const float*)d_in[9];
  const float* wo_b  = (const float*)d_in[10];

  float* out  = (float*)d_out;
  float* attn = out + (size_t)MALL * FDIM;   // [B,h,S,S] region of d_out

  const size_t XE = (size_t)MALL * FDIM;     // 4,718,592 bf16 elements
  const size_t WE = (size_t)FDIM * FDIM;     // 262,144
  unsigned short* Xq  = (unsigned short*)d_ws;
  unsigned short* Xk  = Xq  + XE;
  unsigned short* Xv  = Xk  + XE;
  unsigned short* Qb  = Xv  + XE;            // [b,h,S,d]
  unsigned short* Kb  = Qb  + XE;            // [b,h,S,d]
  unsigned short* Vt  = Kb  + XE;            // [b,h,d,S]
  unsigned short* Ctx = Vt  + XE;            // [B*S, F]
  unsigned short* Wqt = Ctx + XE;
  unsigned short* Wkt = Wqt + WE;
  unsigned short* Wvt = Wkt + WE;
  unsigned short* Wot = Wvt + WE;

  // 1) input + weight conversion
  cvt_f32_bf16<<<(unsigned)(XE / 4 / 256), 256, 0, stream>>>(query, Xq);
  cvt_f32_bf16<<<(unsigned)(XE / 4 / 256), 256, 0, stream>>>(keys,  Xk);
  cvt_f32_bf16<<<(unsigned)(XE / 4 / 256), 256, 0, stream>>>(value, Xv);
  cvt_w_transpose<<<(unsigned)(WE / 256), 256, 0, stream>>>(wq_w, Wqt);
  cvt_w_transpose<<<(unsigned)(WE / 256), 256, 0, stream>>>(wk_w, Wkt);
  cvt_w_transpose<<<(unsigned)(WE / 256), 256, 0, stream>>>(wv_w, Wvt);
  cvt_w_transpose<<<(unsigned)(WE / 256), 256, 0, stream>>>(wo_w, Wot);

  // 2) projections (block tile 128x64, wave tile 16x64)
  dim3 gp(FDIM / 64, MALL / 128);
  gemm_proj<0><<<gp, 256, 0, stream>>>(Xq, Wqt, wq_b, Qb, nullptr);
  gemm_proj<0><<<gp, 256, 0, stream>>>(Xk, Wkt, wk_b, Kb, nullptr);
  gemm_proj<1><<<gp, 256, 0, stream>>>(Xv, Wvt, wv_b, Vt, nullptr);

  // 3) fused logits + softmax + attn-out + ctx (one HBM pass over attn)
  const int SMEM_BYTES = 16 * LDP * 4 + 2048 + 4096;   // 154,112 B (< 320KB/WGP)
  (void)hipFuncSetAttribute(reinterpret_cast<const void*>(attn_fused),
                            hipFuncAttributeMaxDynamicSharedMemorySize, SMEM_BYTES);
  dim3 ga(SEQ / 16, BATCH * NHEADS);
  attn_fused<<<ga, 256, SMEM_BYTES, stream>>>(Qb, Kb, Vt, attn, Ctx);

  // 4) output projection -> d_out
  gemm_proj<2><<<gp, 256, 0, stream>>>(Ctx, Wot, wo_b, nullptr, out);
}
